// TaskAlignedAssigner_18528488915030
// MI455X (gfx1250) — matrix operand order repaired and checked by reading, not compiled, not run on hardware
//
#include <hip/hip_runtime.h>
#include <hip/hip_bf16.h>
#include <math.h>

// ---------------- problem constants (from reference) ----------------
static constexpr int B_ = 16;
static constexpr int A_ = 8400;
static constexpr int C_ = 80;
static constexpr int M_ = 32;
static constexpr int TOPK_ = 13;
static constexpr float EPS_ = 1e-9f;
static constexpr float IOU_EPS_ = 1e-7f;
static constexpr float FOUR_OVER_PI2 = 0.40528473456935109f; // 4/pi^2

static constexpr int NTILES = A_ / 16;            // 525 anchor tiles of 16
static constexpr size_t NBMA = (size_t)B_ * M_ * A_; // 4,300,800 elements per plane

typedef __attribute__((ext_vector_type(16))) _Float16 v16h;
typedef __attribute__((ext_vector_type(8)))  float    v8f;

// ---------------- CIoU (box1 = gt, box2 = pred), masked ----------------
__device__ __forceinline__ float ciou_align(float4 g, float4 p, float ax, float ay,
                                            float mg, float score, float& align_out) {
    // mask_in_gts: anchor strictly inside gt box
    float dmin = fminf(fminf(ax - g.x, ay - g.y), fminf(g.z - ax, g.w - ay));
    float mask = (dmin > EPS_) ? mg : 0.0f;

    float w1 = g.z - g.x, h1 = g.w - g.y + IOU_EPS_;
    float w2 = p.z - p.x, h2 = p.w - p.y + IOU_EPS_;
    float iw = fmaxf(fminf(g.z, p.z) - fmaxf(g.x, p.x), 0.0f);
    float ih = fmaxf(fminf(g.w, p.w) - fmaxf(g.y, p.y), 0.0f);
    float inter = iw * ih;
    float uni = w1 * h1 + w2 * h2 - inter + IOU_EPS_;
    float iou = inter / uni;
    float cw = fmaxf(g.z, p.z) - fminf(g.x, p.x);
    float ch = fmaxf(g.w, p.w) - fminf(g.y, p.y);
    float c2 = cw * cw + ch * ch + IOU_EPS_;
    float dx = p.x + p.z - g.x - g.z;
    float dy = p.y + p.w - g.y - g.w;
    float rho2 = (dx * dx + dy * dy) * 0.25f;
    float da = atanf(w2 / h2) - atanf(w1 / h1);
    float v = FOUR_OVER_PI2 * da * da;
    float alpha = v / (v - iou + (1.0f + IOU_EPS_));
    float ci = iou - (rho2 / c2 + v * alpha);

    float ov = fmaxf(ci, 0.0f) * mask;      // overlaps (masked)
    float sc = score * mask;                // bbox_scores (masked)
    float o2 = ov * ov;
    align_out = sc * (o2 * o2 * o2);        // score^ALPHA * overlap^BETA (1,6)
    return ov;
}

// =====================================================================
// Kernel 1: scores via V_WMMA_F32_16X16X32_F16 (gather-as-GEMM with
// one-hot B built in registers), then masked CIoU + align metric.
// One wave per 16-anchor tile; writes align_metric & overlaps (B,M,A).
// =====================================================================
__global__ __launch_bounds__(256) void k_metric(
    const float* __restrict__ pred_cls, const float* __restrict__ pred_bbox,
    const float* __restrict__ anchor_points, const int* __restrict__ gt_cls,
    const float* __restrict__ gt_bbox, const float* __restrict__ mask_gt,
    float* __restrict__ w_align, float* __restrict__ w_ovl) {
    const int b    = blockIdx.y;
    const int wave = threadIdx.x >> 5;
    const int lane = threadIdx.x & 31;
    const int tile = blockIdx.x * 8 + wave;
    if (tile >= NTILES) return;              // whole-wave uniform exit (EXEC stays full)

    const int a_base = tile * 16;
    const int half   = lane >> 4;            // 0: lanes 0-15, 1: lanes 16-31
    const int li     = lane & 15;

    // per-lane gt data: column m0 (tile 0) and m1 (tile 1)
    const int m0 = li, m1 = 16 + li;
    const float4 g0 = *(const float4*)(gt_bbox + (size_t)(b * M_ + m0) * 4);
    const float4 g1 = *(const float4*)(gt_bbox + (size_t)(b * M_ + m1) * 4);
    const float mg0 = mask_gt[b * M_ + m0];
    const float mg1 = mask_gt[b * M_ + m1];
    const int lab0 = gt_cls[b * M_ + m0];
    const int lab1 = gt_cls[b * M_ + m1];

    // ---- WMMA: scores[a_row, m] = pred_cls[a_row, label[m]] ----
    const int a_row = a_base + li;           // A-fragment row for this lane
    const float* prow = pred_cls + ((size_t)b * A_ + a_row) * C_;
    if (a_row + 16 < A_) __builtin_prefetch(prow + 16 * C_, 0, 0);  // global_prefetch_b8

    v8f acc0 = {}; v8f acc1 = {};
    #pragma unroll
    for (int ks = 0; ks < 3; ++ks) {         // K = 80 padded to 96
        v16h Am;
        const int c0 = ks * 32 + half * 8;         // halves 0-7: K = kb..kb+7
        #pragma unroll
        for (int i = 0; i < 8; ++i) Am[i] = (_Float16)prow[c0 + i];   // c0+7 <= 79 always
        const int c1 = ks * 32 + 16 + half * 8;    // halves 8-15: K = 16+kb..
        if (c1 < C_) {
            #pragma unroll
            for (int i = 0; i < 8; ++i) Am[8 + i] = (_Float16)prow[c1 + i];
        } else {
            #pragma unroll
            for (int i = 0; i < 8; ++i) Am[8 + i] = (_Float16)0.0f;
        }
        // one-hot B fragments: lane col = li, K = (i&8)*2 + half*8 + (i&7) (+ ks*32 global)
        const int base = ks * 32;
        const int r0 = lab0 - base, r1 = lab1 - base;
        v16h B0, B1;
        #pragma unroll
        for (int i = 0; i < 16; ++i) {
            const int kk = (i & 8) * 2 + half * 8 + (i & 7);
            B0[i] = (kk == r0) ? (_Float16)1.0f : (_Float16)0.0f;
            B1[i] = (kk == r1) ? (_Float16)1.0f : (_Float16)0.0f;
        }
        acc0 = __builtin_amdgcn_wmma_f32_16x16x32_f16(false, Am, false, B0, (short)0, acc0, false, false);
        acc1 = __builtin_amdgcn_wmma_f32_16x16x32_f16(false, Am, false, B1, (short)0, acc1, false, false);
    }

    // ---- per-lane CIoU for 8 anchor rows x 2 gt columns ----
    float al0[8], ov0[8], al1[8], ov1[8];
    #pragma unroll
    for (int j = 0; j < 8; ++j) {
        const int a = a_base + half * 8 + j;         // D layout: row = half*8 + j
        const float4 p = *(const float4*)(pred_bbox + ((size_t)b * A_ + a) * 4);
        const float ax = anchor_points[2 * a];
        const float ay = anchor_points[2 * a + 1];
        ov0[j] = ciou_align(g0, p, ax, ay, mg0, acc0[j], al0[j]);
        ov1[j] = ciou_align(g1, p, ax, ay, mg1, acc1[j], al1[j]);
    }

    const size_t i0 = (size_t)(b * M_ + m0) * A_ + a_base + half * 8;
    const size_t i1 = (size_t)(b * M_ + m1) * A_ + a_base + half * 8;
    *(float4*)(w_align + i0)     = make_float4(al0[0], al0[1], al0[2], al0[3]);
    *(float4*)(w_align + i0 + 4) = make_float4(al0[4], al0[5], al0[6], al0[7]);
    *(float4*)(w_align + i1)     = make_float4(al1[0], al1[1], al1[2], al1[3]);
    *(float4*)(w_align + i1 + 4) = make_float4(al1[4], al1[5], al1[6], al1[7]);
    *(float4*)(w_ovl + i0)       = make_float4(ov0[0], ov0[1], ov0[2], ov0[3]);
    *(float4*)(w_ovl + i0 + 4)   = make_float4(ov0[4], ov0[5], ov0[6], ov0[7]);
    *(float4*)(w_ovl + i1)       = make_float4(ov1[0], ov1[1], ov1[2], ov1[3]);
    *(float4*)(w_ovl + i1 + 4)   = make_float4(ov1[4], ov1[5], ov1[6], ov1[7]);
}

// =====================================================================
// Kernel 2: per (b,m) row, top-13 over A via 13 argmax passes in LDS;
// writes mask_topk (1.0 at selected anchors) into w_mp.
// =====================================================================
__global__ __launch_bounds__(256) void k_topk(const float* __restrict__ w_align,
                                              const float* __restrict__ mask_gt,
                                              float* __restrict__ w_mp) {
    const int row = blockIdx.x;              // b*M + m
    const int t = threadIdx.x;
    __shared__ float vals[A_];
    __shared__ float rv[256];
    __shared__ int   ri[256];
    __shared__ int   win[TOPK_];

    const float* src = w_align + (size_t)row * A_;
    float* dst = w_mp + (size_t)row * A_;

    if (mask_gt[row] <= 0.0f) {              // invalid gt: whole row -> 0
        for (int i = t; i < A_; i += 256) dst[i] = 0.0f;
        return;
    }
    for (int i = t; i < A_; i += 256) vals[i] = src[i];
    __syncthreads();

    for (int k = 0; k < TOPK_; ++k) {
        float bv = -1.0f; int bi = 0x7fffffff;
        for (int i = t; i < A_; i += 256) {
            float v = vals[i];
            if (v > bv) { bv = v; bi = i; }
        }
        rv[t] = bv; ri[t] = bi;
        __syncthreads();
        for (int s = 128; s > 0; s >>= 1) {
            if (t < s) {
                float ov = rv[t + s]; int oi = ri[t + s];
                if (ov > rv[t] || (ov == rv[t] && oi < ri[t])) { rv[t] = ov; ri[t] = oi; }
            }
            __syncthreads();
        }
        if (t == 0) { win[k] = ri[0]; vals[ri[0]] = -1.0f; }
        __syncthreads();
    }
    for (int i = t; i < A_; i += 256) {
        float o = 0.0f;
        #pragma unroll
        for (int k = 0; k < TOPK_; ++k) if (win[k] == i) o = 1.0f;
        dst[i] = o;
    }
}

// =====================================================================
// Kernel 3: per (b,a) column: final mask_pos (with multi-assignment fix
// via overlap argmax), fg_mask, target_gt_idx -> bbox/label.
// =====================================================================
__global__ __launch_bounds__(256) void k_resolve(
    const float* __restrict__ anchor_points, const int* __restrict__ gt_cls,
    const float* __restrict__ gt_bbox, const float* __restrict__ mask_gt,
    const float* __restrict__ w_ovl, float* __restrict__ w_mp,
    float* __restrict__ out_fg, float* __restrict__ out_bbox,
    int* __restrict__ w_label) {
    const int b = blockIdx.y;
    const int a = blockIdx.x * 256 + threadIdx.x;
    __shared__ float4 gbox[M_];
    __shared__ float  gmg[M_];
    __shared__ int    glab[M_];
    if (threadIdx.x < M_) {
        int m = threadIdx.x;
        gbox[m] = *(const float4*)(gt_bbox + (size_t)(b * M_ + m) * 4);
        gmg[m]  = mask_gt[b * M_ + m];
        glab[m] = gt_cls[b * M_ + m];
    }
    __syncthreads();
    if (a >= A_) return;

    const float ax = anchor_points[2 * a];
    const float ay = anchor_points[2 * a + 1];
    const size_t colbase = (size_t)b * M_ * A_ + a;

    unsigned bits = 0; int fg = 0;
    for (int m = 0; m < M_; ++m) {
        float mt = w_mp[colbase + (size_t)m * A_];
        float4 g = gbox[m];
        float dmin = fminf(fminf(ax - g.x, ay - g.y), fminf(g.z - ax, g.w - ay));
        if (mt > 0.0f && dmin > EPS_ && gmg[m] > 0.0f) { bits |= (1u << m); fg++; }
    }
    if (fg > 1) {                            // assigned to >1 gt: keep max-overlap gt
        float bv = -1e30f; int bm = 0;
        for (int m = 0; m < M_; ++m) {
            float v = w_ovl[colbase + (size_t)m * A_];
            if (v > bv) { bv = v; bm = m; }
        }
        bits = 1u << bm;
    }
    for (int m = 0; m < M_; ++m)
        w_mp[colbase + (size_t)m * A_] = ((bits >> m) & 1u) ? 1.0f : 0.0f;

    const int tgt = bits ? __builtin_ctz(bits) : 0;   // argmax over m of 0/1 col
    const int oi = b * A_ + a;
    out_fg[oi] = bits ? 1.0f : 0.0f;
    w_label[oi] = glab[tgt];
    *(float4*)(out_bbox + (size_t)oi * 4) = gbox[tgt];
}

// =====================================================================
// Kernel 4: per (b,m): pos_align = max_a(align*mp), pos_overlap = max_a(ovl*mp)
// =====================================================================
__global__ __launch_bounds__(256) void k_rowmax(const float* __restrict__ w_align,
                                                const float* __restrict__ w_ovl,
                                                const float* __restrict__ w_mp,
                                                float* __restrict__ w_pa,
                                                float* __restrict__ w_po) {
    const int row = blockIdx.x;
    const size_t base = (size_t)row * A_;
    __shared__ float sa[256], so[256];
    float ma = 0.0f, mo = 0.0f;              // all candidates >= 0
    for (int i = threadIdx.x; i < A_; i += 256) {
        float mp = w_mp[base + i];
        ma = fmaxf(ma, w_align[base + i] * mp);
        mo = fmaxf(mo, w_ovl[base + i] * mp);
    }
    sa[threadIdx.x] = ma; so[threadIdx.x] = mo;
    __syncthreads();
    for (int s = 128; s > 0; s >>= 1) {
        if (threadIdx.x < s) {
            sa[threadIdx.x] = fmaxf(sa[threadIdx.x], sa[threadIdx.x + s]);
            so[threadIdx.x] = fmaxf(so[threadIdx.x], so[threadIdx.x + s]);
        }
        __syncthreads();
    }
    if (threadIdx.x == 0) { w_pa[row] = sa[0]; w_po[row] = so[0]; }
}

// =====================================================================
// Kernel 5: norm per anchor, scatter one-hot * norm into target_cls.
// =====================================================================
__global__ __launch_bounds__(256) void k_out(const float* __restrict__ w_align,
                                             const float* __restrict__ w_mp,
                                             const float* __restrict__ w_pa,
                                             const float* __restrict__ w_po,
                                             const int* __restrict__ w_label,
                                             const float* __restrict__ out_fg,
                                             float* __restrict__ out_cls) {
    const int b = blockIdx.y;
    const int a = blockIdx.x * 256 + threadIdx.x;
    __shared__ float spa[M_], spo[M_];
    if (threadIdx.x < M_) {
        spa[threadIdx.x] = w_pa[b * M_ + threadIdx.x];
        spo[threadIdx.x] = w_po[b * M_ + threadIdx.x];
    }
    __syncthreads();
    if (a >= A_) return;

    const size_t colbase = (size_t)b * M_ * A_ + a;
    float norm = 0.0f;
    for (int m = 0; m < M_; ++m) {
        float v = w_align[colbase + (size_t)m * A_] * w_mp[colbase + (size_t)m * A_]
                  * spo[m] / (spa[m] + EPS_);
        norm = fmaxf(norm, v);
    }
    const int oi = b * A_ + a;
    const int lab = w_label[oi];
    const float val = (out_fg[oi] > 0.0f) ? norm : 0.0f;
    float* dst = out_cls + (size_t)oi * C_;
    for (int c = 0; c < C_; ++c) dst[c] = (c == lab) ? val : 0.0f;
}

// =====================================================================
extern "C" void kernel_launch(void* const* d_in, const int* in_sizes, int n_in,
                              void* d_out, int out_size, void* d_ws, size_t ws_size,
                              hipStream_t stream) {
    const float* pred_cls      = (const float*)d_in[0];   // (B,A,C) f32
    const float* pred_bbox     = (const float*)d_in[1];   // (B,A,4) f32
    const float* anchor_points = (const float*)d_in[2];   // (A,2)   f32
    const int*   gt_cls        = (const int*)d_in[3];     // (B,M,1) i32
    const float* gt_bbox       = (const float*)d_in[4];   // (B,M,4) f32
    const float* mask_gt       = (const float*)d_in[5];   // (B,M,1) f32

    float* out_cls  = (float*)d_out;                              // (B,A,C)
    float* out_bbox = out_cls + (size_t)B_ * A_ * C_;             // (B,A,4)
    float* out_fg   = out_bbox + (size_t)B_ * A_ * 4;             // (B,A)

    float* w        = (float*)d_ws;
    float* w_align  = w;                     // (B,M,A)
    float* w_ovl    = w + NBMA;              // (B,M,A)
    float* w_mp     = w + 2 * NBMA;          // (B,M,A)
    float* w_pa     = w + 3 * NBMA;          // (B,M)
    float* w_po     = w_pa + B_ * M_;        // (B,M)
    int*   w_label  = (int*)(w_po + B_ * M_);// (B,A)

    dim3 blk(256);
    k_metric<<<dim3((NTILES + 7) / 8, B_), blk, 0, stream>>>(
        pred_cls, pred_bbox, anchor_points, gt_cls, gt_bbox, mask_gt, w_align, w_ovl);
    k_topk<<<dim3(B_ * M_), blk, 0, stream>>>(w_align, mask_gt, w_mp);
    k_resolve<<<dim3((A_ + 255) / 256, B_), blk, 0, stream>>>(
        anchor_points, gt_cls, gt_bbox, mask_gt, w_ovl, w_mp, out_fg, out_bbox, w_label);
    k_rowmax<<<dim3(B_ * M_), blk, 0, stream>>>(w_align, w_ovl, w_mp, w_pa, w_po);
    k_out<<<dim3((A_ + 255) / 256, B_), blk, 0, stream>>>(
        w_align, w_mp, w_pa, w_po, w_label, out_fg, out_cls);
}
